// GCN_10213432229995
// MI455X (gfx1250) — compile-verified, matrix operand order checked by reading, not run on hardware
//
#include <hip/hip_runtime.h>
#include <hip/hip_bf16.h>

typedef __attribute__((ext_vector_type(16))) _Float16 v16h;
typedef __attribute__((ext_vector_type(8)))  _Float16 v8h;
typedef __attribute__((ext_vector_type(8)))  float    v8f;

#define CAT16(lo, hi) __builtin_shufflevector(lo, hi, 0, 1, 2, 3, 4, 5, 6, 7, 8, 9, 10, 11, 12, 13, 14, 15)

// ---------------------------------------------------------------------------
// Small utility kernels
// ---------------------------------------------------------------------------
__global__ void k_fill(float* __restrict__ p, int n, float v) {
  int i = blockIdx.x * blockDim.x + threadIdx.x;
  if (i < n) p[i] = v;
}

__global__ void k_deg(const int* __restrict__ col, float* __restrict__ deg, int E) {
  int e = blockIdx.x * blockDim.x + threadIdx.x;
  if (e < E) atomicAdd(&deg[col[e]], 1.0f);
}

__global__ void k_dinv(float* __restrict__ d, int n) {
  int i = blockIdx.x * blockDim.x + threadIdx.x;
  if (i < n) {
    float v = d[i];
    d[i] = (v > 0.0f) ? rsqrtf(v) : 0.0f;
  }
}

// h[i,:] = hw[i,:] * dinv[i]^2 + bias[:]   (self-loop contribution + bias), float4
__global__ void k_aggr_init(const float* __restrict__ hw, const float* __restrict__ dinv,
                            const float* __restrict__ bias, float* __restrict__ h, int N) {
  int idx = blockIdx.x * blockDim.x + threadIdx.x;   // over N*16 float4s
  if (idx < N * 16) {
    int i = idx >> 4;
    int q = (idx & 15) * 4;
    float s = dinv[i];
    float s2 = s * s;
    const float4 v = *(const float4*)(hw + (size_t)idx * 4);
    float4 o;
    o.x = v.x * s2 + bias[q + 0];
    o.y = v.y * s2 + bias[q + 1];
    o.z = v.z * s2 + bias[q + 2];
    o.w = v.w * s2 + bias[q + 3];
    *(float4*)(h + (size_t)idx * 4) = o;
  }
}

// h[col,:] += hw[row,:] * dinv[row]*dinv[col]   (edge messages), float4 gather
__global__ void k_aggr_edges(const float* __restrict__ hw, const float* __restrict__ dinv,
                             const int* __restrict__ row, const int* __restrict__ col,
                             float* __restrict__ h, int E) {
  long long idx = (long long)blockIdx.x * blockDim.x + threadIdx.x;  // over E*16
  if (idx < (long long)E * 16) {
    int e = (int)(idx >> 4);
    int q = (int)(idx & 15) * 4;
    int r = row[e], c = col[e];
    float w = dinv[r] * dinv[c];
    const float4 m = *(const float4*)(hw + (size_t)r * 64 + q);
    float* dst = h + (size_t)c * 64 + q;
    atomicAdd(dst + 0, m.x * w);
    atomicAdd(dst + 1, m.y * w);
    atomicAdd(dst + 2, m.z * w);
    atomicAdd(dst + 3, m.w * w);
  }
}

// ---------------------------------------------------------------------------
// WMMA GEMM:  out[N,64] = (RELU_IN ? relu(in) : in)[N,64] @ W[64,64]
// Block = 128 threads (4 waves) = 64 output rows.
// Activations (64x64) and W^T (64x64) staged in LDS as f16 via coalesced
// float4 loads. f16 A-fragment layout is K-contiguous in runs of 8, so each
// fragment is two aligned 16B LDS reads; W stored transposed gives B the
// same property.
// ---------------------------------------------------------------------------
template <bool RELU_IN>
__global__ void k_gemm64(const float* __restrict__ in, const float* __restrict__ W,
                         float* __restrict__ out, int N) {
  __shared__ __align__(16) _Float16 sA[64 * 64];   // activation tile, row-major
  __shared__ __align__(16) _Float16 sWT[64 * 64];  // W transposed: sWT[n*64 + k]

  const int tid = threadIdx.x;            // 0..127
  const int rowBase0 = blockIdx.x * 64;

  // Stage A tile (64 rows x 64 cols), ReLU folded into the f32->f16 convert.
#pragma unroll
  for (int i = 0; i < 8; ++i) {
    int fid = i * 128 + tid;              // 0..1023 float4s
    int rrow = fid >> 4;
    int c4 = (fid & 15) * 4;
    int gr = rowBase0 + rrow;
    if (gr > N - 1) gr = N - 1;           // clamp; stores guarded later
    const float4 v = *(const float4*)(in + (size_t)gr * 64 + c4);
    float x0 = v.x, x1 = v.y, x2 = v.z, x3 = v.w;
    if (RELU_IN) {
      x0 = fmaxf(x0, 0.f); x1 = fmaxf(x1, 0.f);
      x2 = fmaxf(x2, 0.f); x3 = fmaxf(x3, 0.f);
    }
    _Float16* d = sA + rrow * 64 + c4;
    d[0] = (_Float16)x0; d[1] = (_Float16)x1;
    d[2] = (_Float16)x2; d[3] = (_Float16)x3;
  }
  // Stage W transposed (64x64).
#pragma unroll
  for (int i = 0; i < 8; ++i) {
    int fid = i * 128 + tid;
    int k = fid >> 4;
    int c4 = (fid & 15) * 4;
    const float4 v = *(const float4*)(W + (size_t)k * 64 + c4);
    sWT[(c4 + 0) * 64 + k] = (_Float16)v.x;
    sWT[(c4 + 1) * 64 + k] = (_Float16)v.y;
    sWT[(c4 + 2) * 64 + k] = (_Float16)v.z;
    sWT[(c4 + 3) * 64 + k] = (_Float16)v.w;
  }
  __syncthreads();

  const int lane = tid & 31;
  const int wave = tid >> 5;
  const int half = lane >> 4;
  const int mn = lane & 15;
  const int rowBase = rowBase0 + wave * 16;

  const v8h* arow = (const v8h*)(sA + (wave * 16 + mn) * 64);
  const v16h a0 = CAT16(arow[half + 0], arow[half + 2]);
  const v16h a1 = CAT16(arow[half + 4], arow[half + 6]);

#pragma unroll
  for (int ct = 0; ct < 4; ++ct) {
    const int colBase = ct * 16;
    const v8h* brow = (const v8h*)(sWT + (colBase + mn) * 64);
    const v16h b0 = CAT16(brow[half + 0], brow[half + 2]);
    const v16h b1 = CAT16(brow[half + 4], brow[half + 6]);
    v8f c = {};
    c = __builtin_amdgcn_wmma_f32_16x16x32_f16(false, a0, false, b0, (short)0, c, false, false);
    c = __builtin_amdgcn_wmma_f32_16x16x32_f16(false, a1, false, b1, (short)0, c, false, false);
#pragma unroll
    for (int r8 = 0; r8 < 8; ++r8) {
      int r = rowBase + half * 8 + r8;
      if (r < N) out[(size_t)r * 64 + colBase + mn] = c[r8];
    }
  }
}

// ---------------------------------------------------------------------------
// Fused MLP head: out[i] = relu(relu(h[i,:]) @ fc1_w + fc1_b) @ fc2_w + fc2_b
// fc1_w (64x256) staged transposed in LDS (32KB f16); the [N,256]
// intermediate lives only in registers. Cross-lane shfl_xor reduces the N
// dimension of each C tile down to one scalar per node.
// ---------------------------------------------------------------------------
__global__ void k_fc_fused(const float* __restrict__ h, const float* __restrict__ fw1,
                           const float* __restrict__ fb1, const float* __restrict__ fw2,
                           const float* __restrict__ fb2, float* __restrict__ out, int N) {
  __shared__ __align__(16) _Float16 sA[64 * 64];     // relu(h) tile
  __shared__ __align__(16) _Float16 sWT[256 * 64];   // fc1_w transposed: [n*64 + k]

  const int tid = threadIdx.x;            // 0..127
  const int rowBase0 = blockIdx.x * 64;

#pragma unroll
  for (int i = 0; i < 8; ++i) {
    int fid = i * 128 + tid;
    int rrow = fid >> 4;
    int c4 = (fid & 15) * 4;
    int gr = rowBase0 + rrow;
    if (gr > N - 1) gr = N - 1;
    const float4 v = *(const float4*)(h + (size_t)gr * 64 + c4);
    _Float16* d = sA + rrow * 64 + c4;
    d[0] = (_Float16)fmaxf(v.x, 0.f); d[1] = (_Float16)fmaxf(v.y, 0.f);
    d[2] = (_Float16)fmaxf(v.z, 0.f); d[3] = (_Float16)fmaxf(v.w, 0.f);
  }
  // fc1_w is 64x256: 4096 float4s -> 32 iterations of 128 threads.
#pragma unroll 8
  for (int i = 0; i < 32; ++i) {
    int fid = i * 128 + tid;
    int k = fid >> 6;                     // row of fc1_w (K dim)
    int c4 = (fid & 63) * 4;              // column (N dim)
    const float4 v = *(const float4*)(fw1 + (size_t)k * 256 + c4);
    sWT[(c4 + 0) * 64 + k] = (_Float16)v.x;
    sWT[(c4 + 1) * 64 + k] = (_Float16)v.y;
    sWT[(c4 + 2) * 64 + k] = (_Float16)v.z;
    sWT[(c4 + 3) * 64 + k] = (_Float16)v.w;
  }
  __syncthreads();

  const int lane = tid & 31;
  const int wave = tid >> 5;
  const int half = lane >> 4;
  const int mn = lane & 15;
  const int rowBase = rowBase0 + wave * 16;

  const v8h* arow = (const v8h*)(sA + (wave * 16 + mn) * 64);
  const v16h a0 = CAT16(arow[half + 0], arow[half + 2]);
  const v16h a1 = CAT16(arow[half + 4], arow[half + 6]);

  float acc[8];
#pragma unroll
  for (int r8 = 0; r8 < 8; ++r8) acc[r8] = 0.f;

#pragma unroll 4
  for (int ct = 0; ct < 16; ++ct) {
    const int colBase = ct * 16;
    const v8h* brow = (const v8h*)(sWT + (colBase + mn) * 64);
    const v16h b0 = CAT16(brow[half + 0], brow[half + 2]);
    const v16h b1 = CAT16(brow[half + 4], brow[half + 6]);
    v8f c = {};
    c = __builtin_amdgcn_wmma_f32_16x16x32_f16(false, a0, false, b0, (short)0, c, false, false);
    c = __builtin_amdgcn_wmma_f32_16x16x32_f16(false, a1, false, b1, (short)0, c, false, false);
    const float bias = fb1[colBase + mn];
    const float wv = fw2[colBase + mn];
#pragma unroll
    for (int r8 = 0; r8 < 8; ++r8) {
      float v = fmaxf(c[r8] + bias, 0.f);
      acc[r8] += v * wv;
    }
  }

  // Reduce across the 16 lanes of each half (the N dimension of the C tile).
#pragma unroll
  for (int r8 = 0; r8 < 8; ++r8) {
    float v = acc[r8];
    v += __shfl_xor(v, 8, 32);
    v += __shfl_xor(v, 4, 32);
    v += __shfl_xor(v, 2, 32);
    v += __shfl_xor(v, 1, 32);
    acc[r8] = v;
  }
  if (mn == 0) {
    const float b2v = fb2[0];
#pragma unroll
    for (int r8 = 0; r8 < 8; ++r8) {
      int r = rowBase + half * 8 + r8;
      if (r < N) out[r] = acc[r8] + b2v;
    }
  }
}

// ---------------------------------------------------------------------------
// Column L2 normalization
// ---------------------------------------------------------------------------
__global__ void k_sumsq(const float* __restrict__ v, float* __restrict__ ss, int N) {
  int i = blockIdx.x * blockDim.x + threadIdx.x;
  float x = (i < N) ? v[i] : 0.f;
  float s = x * x;
  s += __shfl_xor(s, 16, 32);
  s += __shfl_xor(s, 8, 32);
  s += __shfl_xor(s, 4, 32);
  s += __shfl_xor(s, 2, 32);
  s += __shfl_xor(s, 1, 32);
  if ((threadIdx.x & 31) == 0) atomicAdd(ss, s);
}

__global__ void k_norm(float* __restrict__ v, const float* __restrict__ ss, int N) {
  int i = blockIdx.x * blockDim.x + threadIdx.x;
  if (i < N) {
    float s = fmaxf(sqrtf(ss[0]), 1e-12f);
    v[i] = v[i] / s;
  }
}

// ---------------------------------------------------------------------------
extern "C" void kernel_launch(void* const* d_in, const int* in_sizes, int n_in,
                              void* d_out, int out_size, void* d_ws, size_t ws_size,
                              hipStream_t stream) {
  const float* x     = (const float*)d_in[0];
  const int*   ei    = (const int*)d_in[1];
  const float* W1    = (const float*)d_in[2];
  const float* b1    = (const float*)d_in[3];
  const float* W2    = (const float*)d_in[4];
  const float* b2    = (const float*)d_in[5];
  const float* fc1_w = (const float*)d_in[6];
  const float* fc1_b = (const float*)d_in[7];
  const float* fc2_w = (const float*)d_in[8];
  const float* fc2_b = (const float*)d_in[9];

  const int N = in_sizes[0] / 64;
  const int E = in_sizes[1] / 2;
  const int* row = ei;        // source nodes
  const int* col = ei + E;    // target nodes

  float* ws   = (float*)d_ws;
  float* dinv = ws;                          // N floats
  float* ss   = ws + N;                      // 1 float
  float* bufA = ws + N + 16;                 // N*64 floats
  float* bufB = bufA + (size_t)N * 64;       // N*64 floats
  float* out  = (float*)d_out;               // N floats

  const int TB = 256;
  const int nBlocksN   = (N + TB - 1) / TB;
  const int nBlocksE   = (E + TB - 1) / TB;
  const int nBlocksN4  = (int)(((long long)N * 16 + TB - 1) / TB);
  const int nBlocksE4  = (int)(((long long)E * 16 + TB - 1) / TB);
  const int gemmBlocks = (N + 63) / 64;      // 4 waves/block * 16 rows/wave

  // degrees -> dinv (self-loop counts as 1)
  k_fill<<<nBlocksN, TB, 0, stream>>>(dinv, N, 1.0f);
  k_fill<<<1, 32, 0, stream>>>(ss, 1, 0.0f);
  k_deg<<<nBlocksE, TB, 0, stream>>>(col, dinv, E);
  k_dinv<<<nBlocksN, TB, 0, stream>>>(dinv, N);

  // layer 1: bufA = x @ W1 ; bufB = D^-1/2 (A+I) D^-1/2 bufA + b1
  k_gemm64<false><<<gemmBlocks, 128, 0, stream>>>(x, W1, bufA, N);
  k_aggr_init<<<nBlocksN4, TB, 0, stream>>>(bufA, dinv, b1, bufB, N);
  k_aggr_edges<<<nBlocksE4, TB, 0, stream>>>(bufA, dinv, row, col, bufB, E);

  // layer 2 (ReLU of layer-1 output applied during LDS staging)
  k_gemm64<true><<<gemmBlocks, 128, 0, stream>>>(bufB, W2, bufA, N);
  k_aggr_init<<<nBlocksN4, TB, 0, stream>>>(bufA, dinv, b2, bufB, N);
  k_aggr_edges<<<nBlocksE4, TB, 0, stream>>>(bufA, dinv, row, col, bufB, E);

  // fused MLP head (ReLU on staging, fc1+fc2 fused) -> out
  k_fc_fused<<<gemmBlocks, 128, 0, stream>>>(bufB, fc1_w, fc1_b, fc2_w, fc2_b, out, N);

  // column-wise L2 normalize in place
  k_sumsq<<<nBlocksN, TB, 0, stream>>>(out, ss, N);
  k_norm<<<nBlocksN, TB, 0, stream>>>(out, ss, N);
}
